// VQVAE_60413009986017
// MI455X (gfx1250) — matrix operand order repaired and checked by reading, not compiled, not run on hardware
//
#include <hip/hip_runtime.h>
#include <hip/hip_bf16.h>
#include <math.h>

typedef __bf16 bf16_t;
typedef __attribute__((ext_vector_type(16))) __bf16 v16bf;
typedef __attribute__((ext_vector_type(8)))  float  v8f;
typedef unsigned int u32x4 __attribute__((ext_vector_type(4)));
typedef int          i32x8 __attribute__((ext_vector_type(8)));
typedef int          i32x4 __attribute__((ext_vector_type(4)));

union Frag16 { v16bf v; uint4 u[2]; };

static constexpr int EPI_RELU = 1;  // out = bf16 relu(acc+bias)
static constexpr int EPI_Z    = 2;  // out = f32 (acc+bias) AND bf16 copy
static constexpr int EPI_TANH = 3;  // out = f32 tanh(acc+bias)

// TDM availability (device pass only; host pass type-checks the fallback).
#ifdef __HIP_DEVICE_COMPILE__
#  if __has_builtin(__builtin_amdgcn_tensor_load_to_lds)
#    define HAVE_TDM 1
#  else
#    define HAVE_TDM 0
#  endif
#else
#  define HAVE_TDM 0
#endif

#if HAVE_TDM
// Issue a 2D TDM tile load: tile_d1 rows x tile_d0 elems (bf16) from a
// row-major tensor (row stride = stride elems) into LDS at lds_off, with
// optional LDS row padding (interval/amount codes per D# group1 spec).
// This toolchain's builtin takes 6 args (g0, g1, g2, g3, extra, cpol).
__device__ __forceinline__ void tdm_load_2d(
    unsigned lds_off, const void* gptr,
    unsigned tile_d0, unsigned tile_d1,
    unsigned stride, unsigned tens_d0, unsigned tens_d1,
    int pad_interval_code, int pad_amount_code)
{
  unsigned long long ga = (unsigned long long)gptr;
  u32x4 g0;
  g0[0] = 1u;                                               // count=1 (valid)
  g0[1] = lds_off;                                          // lds_addr
  g0[2] = (unsigned)(ga & 0xFFFFFFFFull);                   // global_addr lo
  g0[3] = (unsigned)((ga >> 32) & 0x1FFFFFFull) | (2u << 30); // addr hi | type=2
  unsigned w0 = (1u << 16);                                 // data_size=1 (2B)
  if (pad_amount_code >= 0)
    w0 |= (1u << 20) | ((unsigned)pad_interval_code << 22)
        | ((unsigned)pad_amount_code << 25);
  i32x8 g1;
  g1[0] = (int)w0;
  g1[1] = (int)((tens_d0 & 0xFFFFu) << 16);                 // tensor_dim0 lo
  g1[2] = (int)(((tens_d0 >> 16) & 0xFFFFu) | ((tens_d1 & 0xFFFFu) << 16));
  g1[3] = (int)(((tens_d1 >> 16) & 0xFFFFu) | ((tile_d0 & 0xFFFFu) << 16));
  g1[4] = (int)(tile_d1 & 0xFFFFu);                         // tile_dim1, dim2=0
  g1[5] = (int)stride;                                      // dim0_stride lo32
  g1[6] = 0;
  g1[7] = 0;
  i32x4 z4 = {0, 0, 0, 0};
  i32x8 z8 = {0, 0, 0, 0, 0, 0, 0, 0};
  __builtin_amdgcn_tensor_load_to_lds(g0, g1, z4, z4, z8, 0);
}
#endif

// ---------------------------------------------------------------------------
// Layer-1 GEMM (A is f32 x): manual staging with on-the-fly f32->bf16 convert,
// plus global_prefetch of the next K tile.  out = relu(A@W + b) in bf16.
// Block tile 128x64, BK=32, 256 threads (8 waves), wave tile 32x32 (2x2 WMMA).
// ---------------------------------------------------------------------------
template<int EPI>
__global__ __launch_bounds__(256) void gemm_f32a_kernel(
    const float* __restrict__ A, const bf16_t* __restrict__ Wt,
    const float* __restrict__ bias,
    float* __restrict__ outF, bf16_t* __restrict__ outB,
    int M, int N, int K)
{
  constexpr int BM = 128, BN = 64, BK = 32, LDSD = 40;
  __shared__ __align__(16) bf16_t As[BM][LDSD];
  __shared__ __align__(16) bf16_t Bs[BN][LDSD];

  const int tid  = threadIdx.x;
  const int lane = tid & 31;
  const int w    = tid >> 5;
  const int wm   = w >> 1, wn = w & 1;
  const int bm0  = blockIdx.y * BM;
  const int bn0  = blockIdx.x * BN;

  v8f acc[2][2] = {};

  const int arow = tid >> 1, akc = (tid & 1) * 16;
  const int brow = tid >> 2, bkc = (tid & 3) * 8;

  for (int k0 = 0; k0 < K; k0 += BK) {
    __syncthreads();
    {
      const float* abase = A + (size_t)(bm0 + arow) * K + k0 + akc;
      const float4* src = reinterpret_cast<const float4*>(abase);
      __align__(16) bf16_t tmp[16];
      #pragma unroll
      for (int i = 0; i < 4; ++i) {
        float4 f = src[i];
        tmp[i*4+0] = (bf16_t)f.x; tmp[i*4+1] = (bf16_t)f.y;
        tmp[i*4+2] = (bf16_t)f.z; tmp[i*4+3] = (bf16_t)f.w;
      }
      *reinterpret_cast<uint4*>(&As[arow][akc])     = reinterpret_cast<uint4*>(tmp)[0];
      *reinterpret_cast<uint4*>(&As[arow][akc + 8]) = reinterpret_cast<uint4*>(tmp)[1];
      if (k0 + BK < K) __builtin_prefetch((const void*)(abase + BK), 0, 0);
    }
    {
      *reinterpret_cast<uint4*>(&Bs[brow][bkc]) =
          *reinterpret_cast<const uint4*>(Wt + (size_t)(bn0 + brow) * K + k0 + bkc);
    }
    __syncthreads();

    const int ar = lane & 15, akb = (lane >> 4) * 8, bkb = (lane >> 4) * 16;
    Frag16 a[2], b[2];
    #pragma unroll
    for (int mi = 0; mi < 2; ++mi) {
      const int r = wm * 32 + mi * 16 + ar;
      a[mi].u[0] = *reinterpret_cast<const uint4*>(&As[r][akb]);
      a[mi].u[1] = *reinterpret_cast<const uint4*>(&As[r][akb + 16]);
    }
    #pragma unroll
    for (int ni = 0; ni < 2; ++ni) {
      const int c = wn * 32 + ni * 16 + ar;
      b[ni].u[0] = *reinterpret_cast<const uint4*>(&Bs[c][akb]);
      b[ni].u[1] = *reinterpret_cast<const uint4*>(&Bs[c][akb + 16]);
    }
    #pragma unroll
    for (int mi = 0; mi < 2; ++mi)
      #pragma unroll
      for (int ni = 0; ni < 2; ++ni)
        acc[mi][ni] = __builtin_amdgcn_wmma_f32_16x16x32_bf16(
            false, a[mi].v, false, b[ni].v, (short)0, acc[mi][ni], false, false);
  }

  const int ml = (lane >> 4) * 8, nl = lane & 15;
  #pragma unroll
  for (int mi = 0; mi < 2; ++mi)
    #pragma unroll
    for (int ni = 0; ni < 2; ++ni) {
      const int col = bn0 + wn * 32 + ni * 16 + nl;
      const float bv = bias[col];
      #pragma unroll
      for (int r = 0; r < 8; ++r) {
        const int row = bm0 + wm * 32 + mi * 16 + ml + r;
        float v = acc[mi][ni][r] + bv;
        if (EPI == EPI_RELU) {
          v = v > 0.f ? v : 0.f;
          outB[(size_t)row * N + col] = (bf16_t)v;
        } else if (EPI == EPI_Z) {
          outF[(size_t)row * N + col] = v;
          outB[(size_t)row * N + col] = (bf16_t)v;
        } else {
          outF[(size_t)row * N + col] = tanhf(v);
        }
      }
    }
}

// ---------------------------------------------------------------------------
// bf16-A GEMM with TDM double buffering: wave 0 issues tensor_load_to_lds for
// the next k-step's A/B tiles (pad fields keep the 40-elem LDS row stride)
// while all 8 waves run WMMA on the current buffers.  One barrier per step.
// ---------------------------------------------------------------------------
template<int EPI>
__global__ __launch_bounds__(256) void gemm_bf16_tdm_kernel(
    const bf16_t* __restrict__ A, const bf16_t* __restrict__ Wt,
    const float* __restrict__ bias,
    float* __restrict__ outF, bf16_t* __restrict__ outB,
    int M, int N, int K)
{
  constexpr int BM = 128, BN = 64, BK = 32, LDSD = 40;
  __shared__ __align__(16) bf16_t As[2][BM][LDSD];
  __shared__ __align__(16) bf16_t Bs[2][BN][LDSD];

  const int tid  = threadIdx.x;
  const int lane = tid & 31;
  const int w    = tid >> 5;
  const int wm   = w >> 1, wn = w & 1;
  const int bm0  = blockIdx.y * BM;
  const int bn0  = blockIdx.x * BN;
  const int nsteps = K / BK;

  v8f acc[2][2] = {};

#if HAVE_TDM
  // 32 bf16 per row = 16 DWORDs -> pad_interval code 3; pad 4 DWORDs -> code 3
  if (tid < 32) {
    tdm_load_2d((unsigned)(size_t)&As[0][0][0], A + (size_t)bm0 * K,
                BK, BM, (unsigned)K, (unsigned)K, (unsigned)M, 3, 3);
    tdm_load_2d((unsigned)(size_t)&Bs[0][0][0], Wt + (size_t)bn0 * K,
                BK, BN, (unsigned)K, (unsigned)K, (unsigned)N, 3, 3);
  }
#else
  const int arow = tid >> 1, akc = (tid & 1) * 16;
  const int brow = tid >> 2, bkc = (tid & 3) * 8;
#endif

  for (int s = 0; s < nsteps; ++s) {
    const int cur = s & 1;
#if HAVE_TDM
    if (tid < 32) __builtin_amdgcn_s_wait_tensorcnt(0);
    __syncthreads();                       // tiles visible; prev step done
    if (tid < 32 && s + 1 < nsteps) {
      const int k1 = (s + 1) * BK;
      tdm_load_2d((unsigned)(size_t)&As[cur ^ 1][0][0],
                  A + (size_t)bm0 * K + k1,
                  BK, BM, (unsigned)K, (unsigned)K, (unsigned)M, 3, 3);
      tdm_load_2d((unsigned)(size_t)&Bs[cur ^ 1][0][0],
                  Wt + (size_t)bn0 * K + k1,
                  BK, BN, (unsigned)K, (unsigned)K, (unsigned)N, 3, 3);
    }
#else
    const int k0 = s * BK;
    __syncthreads();
    *reinterpret_cast<uint4*>(&As[cur][arow][akc]) =
        *reinterpret_cast<const uint4*>(A + (size_t)(bm0 + arow) * K + k0 + akc);
    *reinterpret_cast<uint4*>(&As[cur][arow][akc + 8]) =
        *reinterpret_cast<const uint4*>(A + (size_t)(bm0 + arow) * K + k0 + akc + 8);
    *reinterpret_cast<uint4*>(&Bs[cur][brow][bkc]) =
        *reinterpret_cast<const uint4*>(Wt + (size_t)(bn0 + brow) * K + k0 + bkc);
    __syncthreads();
#endif

    const int ar = lane & 15, akb = (lane >> 4) * 8, bkb = (lane >> 4) * 16;
    Frag16 a[2], b[2];
    #pragma unroll
    for (int mi = 0; mi < 2; ++mi) {
      const int r = wm * 32 + mi * 16 + ar;
      a[mi].u[0] = *reinterpret_cast<const uint4*>(&As[cur][r][akb]);
      a[mi].u[1] = *reinterpret_cast<const uint4*>(&As[cur][r][akb + 16]);
    }
    #pragma unroll
    for (int ni = 0; ni < 2; ++ni) {
      const int c = wn * 32 + ni * 16 + ar;
      b[ni].u[0] = *reinterpret_cast<const uint4*>(&Bs[cur][c][akb]);
      b[ni].u[1] = *reinterpret_cast<const uint4*>(&Bs[cur][c][akb + 16]);
    }
    #pragma unroll
    for (int mi = 0; mi < 2; ++mi)
      #pragma unroll
      for (int ni = 0; ni < 2; ++ni)
        acc[mi][ni] = __builtin_amdgcn_wmma_f32_16x16x32_bf16(
            false, a[mi].v, false, b[ni].v, (short)0, acc[mi][ni], false, false);
  }

  const int ml = (lane >> 4) * 8, nl = lane & 15;
  #pragma unroll
  for (int mi = 0; mi < 2; ++mi)
    #pragma unroll
    for (int ni = 0; ni < 2; ++ni) {
      const int col = bn0 + wn * 32 + ni * 16 + nl;
      const float bv = bias[col];
      #pragma unroll
      for (int r = 0; r < 8; ++r) {
        const int row = bm0 + wm * 32 + mi * 16 + ml + r;
        float v = acc[mi][ni][r] + bv;
        if (EPI == EPI_RELU) {
          v = v > 0.f ? v : 0.f;
          outB[(size_t)row * N + col] = (bf16_t)v;
        } else if (EPI == EPI_Z) {
          outF[(size_t)row * N + col] = v;
          outB[(size_t)row * N + col] = (bf16_t)v;
        } else {
          outF[(size_t)row * N + col] = tanhf(v);
        }
      }
    }
}

// ---------------------------------------------------------------------------
// Fused VQ kernel: 16 rows of z vs all 1024 codebook entries.  Codebook
// chunks (256x64 bf16, TDM-staged with padded rows) resident in LDS; WMMA
// computes z@e^T; argmin uses e_sq - 2*dot (||z||^2 constant per row).
// Gathers z_q, writes indices, deterministic per-block commitment partial.
// ---------------------------------------------------------------------------
__global__ __launch_bounds__(256) void vq_dist_kernel(
    const bf16_t* __restrict__ zb, const bf16_t* __restrict__ cb,
    const float* __restrict__ cbf, const float* __restrict__ e_sq,
    float* __restrict__ out_idx, bf16_t* __restrict__ zq,
    float* __restrict__ partials, int M)
{
  __shared__ __align__(16) bf16_t zt[16][72];
  __shared__ __align__(16) bf16_t cbt[256][72];
  __shared__ float rv[8][32][8];
  __shared__ int   ri[8][32][8];
  __shared__ float cs[16];

  const int tid  = threadIdx.x;
  const int lane = tid & 31;
  const int w    = tid >> 5;
  const int bm0  = blockIdx.x * 16;

  {
    const int m = tid >> 4, c = (tid & 15) * 4;
    *reinterpret_cast<uint2*>(&zt[m][c]) =
        *reinterpret_cast<const uint2*>(zb + (size_t)(bm0 + m) * 64 + c);
  }

  float bestv[8];
  int   besti[8];
  #pragma unroll
  for (int r = 0; r < 8; ++r) { bestv[r] = 3.0e38f; besti[r] = 0; }

  const int ar = lane & 15, akb = (lane >> 4) * 8, bkb = (lane >> 4) * 16;

  for (int chunk = 0; chunk < 4; ++chunk) {
    __syncthreads();                       // previous chunk's compute done
#if HAVE_TDM
    // 64 bf16/row = 32 DWORDs -> interval code 4; pad 4 DWORDs -> code 3
    if (tid < 32) {
      tdm_load_2d((unsigned)(size_t)&cbt[0][0],
                  cb + (size_t)chunk * 256 * 64,
                  64u, 256u, 64u, 64u, 1024u, 4, 3);
      __builtin_amdgcn_s_wait_tensorcnt(0);
    }
#else
    {
      const bf16_t* src = cb + (size_t)(chunk * 256 + tid) * 64;
      #pragma unroll
      for (int i = 0; i < 8; ++i)
        *reinterpret_cast<uint4*>(&cbt[tid][i * 8]) =
            reinterpret_cast<const uint4*>(src)[i];
    }
#endif
    __syncthreads();

    Frag16 a0, a1;
    a0.u[0] = *reinterpret_cast<const uint4*>(&zt[ar][akb]);
    a0.u[1] = *reinterpret_cast<const uint4*>(&zt[ar][akb + 16]);
    a1.u[0] = *reinterpret_cast<const uint4*>(&zt[ar][32 + akb]);
    a1.u[1] = *reinterpret_cast<const uint4*>(&zt[ar][32 + akb + 16]);

    #pragma unroll
    for (int ni = 0; ni < 2; ++ni) {
      const int c = w * 32 + ni * 16 + ar;
      Frag16 b0, b1;
      b0.u[0] = *reinterpret_cast<const uint4*>(&cbt[c][akb]);
      b0.u[1] = *reinterpret_cast<const uint4*>(&cbt[c][akb + 16]);
      b1.u[0] = *reinterpret_cast<const uint4*>(&cbt[c][32 + akb]);
      b1.u[1] = *reinterpret_cast<const uint4*>(&cbt[c][32 + akb + 16]);
      v8f acc = {};
      acc = __builtin_amdgcn_wmma_f32_16x16x32_bf16(
          false, a0.v, false, b0.v, (short)0, acc, false, false);
      acc = __builtin_amdgcn_wmma_f32_16x16x32_bf16(
          false, a1.v, false, b1.v, (short)0, acc, false, false);
      const int cg = chunk * 256 + c;
      const float esq = e_sq[cg];
      #pragma unroll
      for (int r = 0; r < 8; ++r) {
        const float dv = esq - 2.0f * acc[r];
        if (dv < bestv[r]) { bestv[r] = dv; besti[r] = cg; }
      }
    }
  }

  #pragma unroll
  for (int r = 0; r < 8; ++r) { rv[w][lane][r] = bestv[r]; ri[w][lane][r] = besti[r]; }
  __syncthreads();

  if (tid < 16) {
    const int m = tid, g = m >> 3, s = m & 7;
    float bv = 3.0e38f; int bi = 0;
    for (int wv = 0; wv < 8; ++wv)
      for (int l = 0; l < 16; ++l) {
        const float v = rv[wv][g * 16 + l][s];
        if (v < bv) { bv = v; bi = ri[wv][g * 16 + l][s]; }
      }
    const int row = bm0 + m;
    out_idx[row] = (float)bi;
    float csum = 0.f;
    const float* e = cbf + (size_t)bi * 64;
    for (int d = 0; d < 64; ++d) {
      const float cv = e[d];
      zq[(size_t)row * 64 + d] = (bf16_t)cv;
      const float zv = (float)zt[m][d];
      const float df = cv - zv;
      csum += df * df;
    }
    cs[m] = csum;
  }
  __syncthreads();
  if (tid == 0) {
    float s = 0.f;
    for (int i = 0; i < 16; ++i) s += cs[i];
    partials[blockIdx.x] = s;
  }
}

// ---------------------------------------------------------------------------
// Prep + finalize kernels
// ---------------------------------------------------------------------------
__global__ void wconv_t_kernel(const float* __restrict__ W,
                               bf16_t* __restrict__ Wt, int K, int N) {
  int t = blockIdx.x * blockDim.x + threadIdx.x;
  if (t >= K * N) return;
  int k = t / N, n = t % N;
  Wt[(size_t)n * K + k] = (bf16_t)W[t];
}

__global__ void cbconv_kernel(const float* __restrict__ src,
                              bf16_t* __restrict__ dst, int total) {
  int t = blockIdx.x * blockDim.x + threadIdx.x;
  if (t < total) dst[t] = (bf16_t)src[t];
}

__global__ void esq_kernel(const float* __restrict__ cbf,
                           float* __restrict__ e_sq) {
  int k = blockIdx.x * blockDim.x + threadIdx.x;
  if (k < 1024) {
    float s = 0.f;
    for (int d = 0; d < 64; ++d) { float v = cbf[k * 64 + d]; s += v * v; }
    e_sq[k] = s;
  }
}

__global__ void loss_finalize_kernel(const float* __restrict__ partials,
                                     float* __restrict__ out, int n, float scale) {
  if (threadIdx.x == 0 && blockIdx.x == 0) {
    float s = 0.f;
    for (int i = 0; i < n; ++i) s += partials[i];
    *out = scale * s;
  }
}

// ---------------------------------------------------------------------------
extern "C" void kernel_launch(void* const* d_in, const int* in_sizes, int n_in,
                              void* d_out, int out_size, void* d_ws, size_t ws_size,
                              hipStream_t stream) {
  (void)n_in; (void)out_size; (void)ws_size;
  const float* x    = (const float*)d_in[0];
  const float* We1  = (const float*)d_in[1];
  const float* be1  = (const float*)d_in[2];
  const float* We2  = (const float*)d_in[3];
  const float* be2  = (const float*)d_in[4];
  const float* We3  = (const float*)d_in[5];
  const float* be3  = (const float*)d_in[6];
  const float* cbf  = (const float*)d_in[7];
  const float* Wd1  = (const float*)d_in[8];
  const float* bd1  = (const float*)d_in[9];
  const float* Wd2  = (const float*)d_in[10];
  const float* bd2  = (const float*)d_in[11];
  const float* Wd3  = (const float*)d_in[12];
  const float* bd3  = (const float*)d_in[13];

  const int M = in_sizes[0] / 1024;            // 65536

  char* ws = (char*)d_ws;
  bf16_t* Wt1  = (bf16_t*)(ws + 0);            // 512x1024
  bf16_t* Wt2  = (bf16_t*)(ws + 1048576);      // 256x512
  bf16_t* Wt3  = (bf16_t*)(ws + 1310720);      // 64x256
  bf16_t* Wd1t = (bf16_t*)(ws + 1343488);      // 256x64
  bf16_t* Wd2t = (bf16_t*)(ws + 1376256);      // 512x256
  bf16_t* Wd3t = (bf16_t*)(ws + 1638400);      // 1024x512
  bf16_t* cb_b = (bf16_t*)(ws + 2686976);      // 1024x64
  float*  e_sq = (float*) (ws + 2818048);      // 1024
  float*  lpar = (float*) (ws + 2822144);      // 4096 partials
  bf16_t* h1   = (bf16_t*)(ws + 2838528);
  bf16_t* h2   = (bf16_t*)(ws + 2838528 + (size_t)M * 512 * 2);
  bf16_t* z_b  = (bf16_t*)(ws + 2838528 + (size_t)M * 512 * 2 + (size_t)M * 256 * 2);
  bf16_t* zq   = (bf16_t*)((char*)z_b + (size_t)M * 64 * 2);
  bf16_t* d1   = h2;
  bf16_t* d2   = h1;

  float* outF    = (float*)d_out;
  float* x_rec   = outF;
  float* z_out   = outF + (size_t)M * 1024;
  float* idxout  = outF + (size_t)M * 1024 + (size_t)M * 64;
  float* lossout = idxout + M;

  auto tgrid = [](int n) { return dim3((unsigned)((n + 255) / 256)); };
  wconv_t_kernel<<<tgrid(1024 * 512), 256, 0, stream>>>(We1, Wt1, 1024, 512);
  wconv_t_kernel<<<tgrid(512 * 256),  256, 0, stream>>>(We2, Wt2, 512, 256);
  wconv_t_kernel<<<tgrid(256 * 64),   256, 0, stream>>>(We3, Wt3, 256, 64);
  wconv_t_kernel<<<tgrid(64 * 256),   256, 0, stream>>>(Wd1, Wd1t, 64, 256);
  wconv_t_kernel<<<tgrid(256 * 512),  256, 0, stream>>>(Wd2, Wd2t, 256, 512);
  wconv_t_kernel<<<tgrid(512 * 1024), 256, 0, stream>>>(Wd3, Wd3t, 512, 1024);
  cbconv_kernel<<<tgrid(1024 * 64), 256, 0, stream>>>(cbf, cb_b, 1024 * 64);
  esq_kernel<<<dim3(4), 256, 0, stream>>>(cbf, e_sq);

  const unsigned gy = (unsigned)(M / 128);
  // ---- encoder ----
  gemm_f32a_kernel<EPI_RELU><<<dim3(512 / 64, gy), 256, 0, stream>>>(
      x,  Wt1, be1, nullptr, h1, M, 512, 1024);
  gemm_bf16_tdm_kernel<EPI_RELU><<<dim3(256 / 64, gy), 256, 0, stream>>>(
      h1, Wt2, be2, nullptr, h2, M, 256, 512);
  gemm_bf16_tdm_kernel<EPI_Z><<<dim3(64 / 64, gy), 256, 0, stream>>>(
      h2, Wt3, be3, z_out, z_b, M, 64, 256);

  // ---- VQ ----
  vq_dist_kernel<<<dim3((unsigned)(M / 16)), 256, 0, stream>>>(
      z_b, cb_b, cbf, e_sq, idxout, zq, lpar, M);

  // ---- decoder (forward value of straight-through z is exactly z_q) ----
  gemm_bf16_tdm_kernel<EPI_RELU><<<dim3(256 / 64, gy), 256, 0, stream>>>(
      zq, Wd1t, bd1, nullptr, d1, M, 256, 64);
  gemm_bf16_tdm_kernel<EPI_RELU><<<dim3(512 / 64, gy), 256, 0, stream>>>(
      d1, Wd2t, bd2, nullptr, d2, M, 512, 256);
  gemm_bf16_tdm_kernel<EPI_TANH><<<dim3(1024 / 64, gy), 256, 0, stream>>>(
      d2, Wd3t, bd3, x_rec, nullptr, M, 1024, 512);

  loss_finalize_kernel<<<dim3(1), 64, 0, stream>>>(
      lpar, lossout, M / 16, 0.25f / ((float)M * 64.0f));
}
